// DecoderModule_43722767073775
// MI455X (gfx1250) — compile-verified, matrix operand order checked by reading, not run on hardware
//
#include <hip/hip_runtime.h>
#include <hip/hip_bf16.h>
#include <math.h>

#define NUM_HYPS 65536
#define VOCAB    500
#define DIM      512   // DEC_DIM == JOINER_DIM == 512
#define VPAD     512   // vocab padded to 512 for WMMA tiling
#define MROWS    32    // hyp rows per block (2 x 16-row WMMA M-tiles)

typedef __attribute__((ext_vector_type(16))) __bf16 v16bf;
typedef __attribute__((ext_vector_type(8)))  float  v8f;

struct alignas(16) U4 { unsigned int x, y, z, w; };
union Frag { v16bf v; U4 u[2]; };

__device__ __forceinline__ unsigned short f2bf(float x) {
    unsigned int u = __float_as_uint(x);
    u += 0x7FFFu + ((u >> 16) & 1u);         // round-to-nearest-even
    return (unsigned short)(u >> 16);
}

__device__ __forceinline__ v8f wmma_bf16(const Frag& a, const Frag& b, v8f c) {
    return __builtin_amdgcn_wmma_f32_16x16x32_bf16(
        false, a.v, false, b.v, (short)0, c, false, false);
}

// ---------------------------------------------------------------------------
// Kernel 0: convert proj_w (512x512) and joiner_w (500x512, zero-padded to
// 512x512) from f32 to bf16 once per launch.
// ---------------------------------------------------------------------------
__global__ void convert_w_kernel(const float* __restrict__ pw,
                                 const float* __restrict__ jw,
                                 unsigned short* __restrict__ pwb,
                                 unsigned short* __restrict__ jwb) {
    int idx = blockIdx.x * 256 + threadIdx.x;          // < 512*512
    pwb[idx] = f2bf(pw[idx]);
    int r = idx >> 9, c = idx & 511;
    jwb[idx] = (r < VOCAB) ? f2bf(jw[r * DIM + c]) : (unsigned short)0;
}

// ---------------------------------------------------------------------------
// Kernel 1: decoder front-end: masked embedding gather + grouped conv1d
// (groups=128, in_ch/group=4, k=2) + ReLU, emitted as bf16 (N x 512).
// ---------------------------------------------------------------------------
__global__ void frontend_kernel(const int* __restrict__ din,
                                const float* __restrict__ embed,
                                const float* __restrict__ convw,
                                unsigned short* __restrict__ dec_bf) {
    int idx = blockIdx.x * 256 + threadIdx.x;          // < N*512
    int n = idx >> 9, o = idx & 511;
    int t0 = din[2 * n], t1 = din[2 * n + 1];
    const float* w = convw + o * 8;                    // (o,i,k) -> o*8+i*2+k
    int g4 = (o >> 2) << 2;                            // group input base chan
    float s = 0.f;
    if (t0 >= 0) {
        const float* e = embed + (size_t)t0 * DIM + g4;
        s += e[0]*w[0] + e[1]*w[2] + e[2]*w[4] + e[3]*w[6];
    }
    if (t1 >= 0) {
        const float* e = embed + (size_t)t1 * DIM + g4;
        s += e[0]*w[1] + e[1]*w[3] + e[2]*w[5] + e[3]*w[7];
    }
    dec_bf[idx] = f2bf(fmaxf(s, 0.f));
}

// ---------------------------------------------------------------------------
// Kernel 2 (the big one): per block of 32 hyp rows:
//   GEMM1 (bf16 WMMA) + bias + enc-add + tanh  -> LDS bf16 tile
//   GEMM2 (bf16 WMMA) + bias                   -> LDS f32 logits
//   row-wise log-softmax stats, block top-4 of (log_softmax + hyp_lp)
// Wave tile: 32(M) x 64(N); each B fragment feeds 2 WMMAs (1.5 loads/WMMA).
// ---------------------------------------------------------------------------
__global__ void __launch_bounds__(256)
fused_gemm_topk_kernel(const unsigned short* __restrict__ dec_bf,
                       const unsigned short* __restrict__ pwb,
                       const unsigned short* __restrict__ jwb,
                       const float* __restrict__ proj_b,
                       const float* __restrict__ enc,
                       const float* __restrict__ joiner_b,
                       const float* __restrict__ hlp,
                       float* __restrict__ cand_val,
                       int* __restrict__ cand_idx) {
    __shared__ __align__(16) unsigned short s_a2[MROWS][DIM];  // 32 KB
    __shared__ __align__(16) float s_logits[MROWS][VPAD];      // 64 KB
    __shared__ float s_rowmax[MROWS], s_rowls[MROWS], s_hlp[MROWS];
    __shared__ float s_redv[256];
    __shared__ int   s_redi[256];
    __shared__ float s_winv[4];
    __shared__ int   s_wini[4];

    const int tid  = threadIdx.x;
    const int wv   = tid >> 5;            // wave 0..7
    const int lane = tid & 31;
    const int m0   = blockIdx.x << 5;     // 32 rows per block
    const int nb   = wv << 6;             // 64-col strip per wave
    const int fr   = lane & 15;           // fragment row/col within tile
    const int kh   = (lane >> 4) << 3;    // K half offset (0 or 8)
    const int rbase = (lane >> 4) << 3;   // C/D: M = v + 8*(lane>=16)

    // ---- Stage 1: dec_proj = ReLUconv @ proj_w.T + b; a2 = tanh(enc + dec)
    v8f acc[2][4];
    #pragma unroll
    for (int mt = 0; mt < 2; ++mt)
        #pragma unroll
        for (int j = 0; j < 4; ++j) acc[mt][j] = (v8f){0,0,0,0,0,0,0,0};

    for (int k0 = 0; k0 < DIM; k0 += 32) {
        Frag a[2];
        #pragma unroll
        for (int mt = 0; mt < 2; ++mt) {
            const unsigned short* pa =
                dec_bf + (size_t)(m0 + mt * 16 + fr) * DIM + k0 + kh;
            a[mt].u[0] = *(const U4*)pa;
            a[mt].u[1] = *(const U4*)(pa + 16);
        }
        #pragma unroll
        for (int j = 0; j < 4; ++j) {
            Frag b;
            const unsigned short* pb =
                pwb + (size_t)(nb + j * 16 + fr) * DIM + k0 + kh;
            b.u[0] = *(const U4*)pb;
            b.u[1] = *(const U4*)(pb + 16);
            acc[0][j] = wmma_bf16(a[0], b, acc[0][j]);
            acc[1][j] = wmma_bf16(a[1], b, acc[1][j]);
        }
    }
    #pragma unroll
    for (int j = 0; j < 4; ++j) {
        int col = nb + j * 16 + fr;
        float pbv = proj_b[col];
        #pragma unroll
        for (int mt = 0; mt < 2; ++mt) {
            #pragma unroll
            for (int v = 0; v < 8; ++v) {
                int r = mt * 16 + rbase + v;
                float x = acc[mt][j][v] + pbv + enc[(size_t)(m0 + r) * DIM + col];
                s_a2[r][col] = f2bf(tanhf(x));
            }
        }
    }
    __syncthreads();

    // ---- Stage 2: logits = a2 @ joiner_w.T + joiner_b (vocab padded)
    v8f acc2[2][4];
    #pragma unroll
    for (int mt = 0; mt < 2; ++mt)
        #pragma unroll
        for (int j = 0; j < 4; ++j) acc2[mt][j] = (v8f){0,0,0,0,0,0,0,0};

    for (int k0 = 0; k0 < DIM; k0 += 32) {
        Frag a[2];
        #pragma unroll
        for (int mt = 0; mt < 2; ++mt) {
            const unsigned short* pa = &s_a2[mt * 16 + fr][k0 + kh];
            a[mt].u[0] = *(const U4*)pa;
            a[mt].u[1] = *(const U4*)(pa + 16);
        }
        #pragma unroll
        for (int j = 0; j < 4; ++j) {
            Frag b;
            const unsigned short* pb =
                jwb + (size_t)(nb + j * 16 + fr) * DIM + k0 + kh;
            b.u[0] = *(const U4*)pb;
            b.u[1] = *(const U4*)(pb + 16);
            acc2[0][j] = wmma_bf16(a[0], b, acc2[0][j]);
            acc2[1][j] = wmma_bf16(a[1], b, acc2[1][j]);
        }
    }
    #pragma unroll
    for (int j = 0; j < 4; ++j) {
        int col = nb + j * 16 + fr;
        float jbv = (col < VOCAB) ? joiner_b[col] : 0.f;
        #pragma unroll
        for (int mt = 0; mt < 2; ++mt) {
            #pragma unroll
            for (int v = 0; v < 8; ++v) {
                int r = mt * 16 + rbase + v;
                s_logits[r][col] = (col < VOCAB) ? acc2[mt][j][v] + jbv : -1e30f;
            }
        }
    }
    if (tid < MROWS) s_hlp[tid] = hlp[m0 + tid];
    __syncthreads();

    // ---- Stage 3: per-row log-softmax stats (4 rows per wave)
    #pragma unroll
    for (int rr = 0; rr < 4; ++rr) {
        int r = wv + rr * 8;
        float m = -1e30f;
        for (int c = lane; c < VOCAB; c += 32) m = fmaxf(m, s_logits[r][c]);
        #pragma unroll
        for (int o2 = 16; o2 > 0; o2 >>= 1) m = fmaxf(m, __shfl_xor(m, o2, 32));
        float su = 0.f;
        for (int c = lane; c < VOCAB; c += 32) su += expf(s_logits[r][c] - m);
        #pragma unroll
        for (int o2 = 16; o2 > 0; o2 >>= 1) su += __shfl_xor(su, o2, 32);
        if (lane == 0) { s_rowmax[r] = m; s_rowls[r] = logf(su); }
    }
    __syncthreads();

    // ---- Stage 4: block-local top-4 of (log_softmax + hyp_log_prob)
    int ex0 = -1, ex1 = -1, ex2 = -1, ex3 = -1;
    for (int round = 0; round < 4; ++round) {
        float bv = -1e30f; int bi = 0x7fffffff;
        for (int f = tid; f < MROWS * VPAD; f += 256) {
            int c = f & 511;
            if (c >= VOCAB) continue;
            if (f == ex0 || f == ex1 || f == ex2 || f == ex3) continue;
            int r = f >> 9;
            float val = s_logits[r][c] - s_rowmax[r] - s_rowls[r] + s_hlp[r];
            if (val > bv || (val == bv && f < bi)) { bv = val; bi = f; }
        }
        s_redv[tid] = bv; s_redi[tid] = bi;
        __syncthreads();
        for (int s = 128; s > 0; s >>= 1) {
            if (tid < s) {
                float ov = s_redv[tid + s]; int oi = s_redi[tid + s];
                if (ov > s_redv[tid] || (ov == s_redv[tid] && oi < s_redi[tid])) {
                    s_redv[tid] = ov; s_redi[tid] = oi;
                }
            }
            __syncthreads();
        }
        if (tid == 0) { s_winv[round] = s_redv[0]; s_wini[round] = s_redi[0]; }
        __syncthreads();
        int w = s_wini[round];
        if (round == 0) ex0 = w; else if (round == 1) ex1 = w;
        else if (round == 2) ex2 = w; else ex3 = w;
    }
    if (tid < 4) {
        int f = s_wini[tid];
        int r = f >> 9, c = f & 511;
        cand_val[blockIdx.x * 4 + tid] = s_winv[tid];
        cand_idx[blockIdx.x * 4 + tid] = (m0 + r) * VOCAB + c;
    }
}

// ---------------------------------------------------------------------------
// Kernel 3: reduce (N/32)*4 candidates to global top-4 and emit outputs.
// ---------------------------------------------------------------------------
__global__ void final_topk_kernel(const float* __restrict__ cand_val,
                                  const int* __restrict__ cand_idx,
                                  const float* __restrict__ hlp,
                                  float* __restrict__ out) {
    __shared__ float rv[256];
    __shared__ int   ri[256];
    __shared__ float wv_[4];
    __shared__ int   wi_[4];
    const int tid = threadIdx.x;
    const int NC = (NUM_HYPS / MROWS) * 4;

    int ex0 = -1, ex1 = -1, ex2 = -1, ex3 = -1;   // exclusion by flat index
    for (int round = 0; round < 4; ++round) {
        float bv = -1e30f; int bfi = 0x7fffffff;
        for (int p = tid; p < NC; p += 256) {
            int fi = cand_idx[p];
            if (fi == ex0 || fi == ex1 || fi == ex2 || fi == ex3) continue;
            float v = cand_val[p];
            if (v > bv || (v == bv && fi < bfi)) { bv = v; bfi = fi; }
        }
        rv[tid] = bv; ri[tid] = bfi;
        __syncthreads();
        for (int s = 128; s > 0; s >>= 1) {
            if (tid < s) {
                float ov = rv[tid + s]; int oi = ri[tid + s];
                if (ov > rv[tid] || (ov == rv[tid] && oi < ri[tid])) {
                    rv[tid] = ov; ri[tid] = oi;
                }
            }
            __syncthreads();
        }
        if (tid == 0) { wv_[round] = rv[0]; wi_[round] = ri[0]; }
        __syncthreads();
        int w = wi_[round];
        if (round == 0) ex0 = w; else if (round == 1) ex1 = w;
        else if (round == 2) ex2 = w; else ex3 = w;
    }
    if (tid == 0) {
        for (int j = 0; j < 4; ++j) {
            float v = wv_[j]; int fi = wi_[j];
            int hyp = fi / VOCAB, tok = fi - hyp * VOCAB;
            out[j]      = v;                      // hyps_topk_log_prob
            out[4 + j]  = expf(v - hlp[hyp]);     // tokens_topk_prob
            out[8 + j]  = (float)hyp;             // topk_hyp_indexes
            out[12 + j] = (float)tok;             // topk_token_indexes
        }
    }
}

extern "C" void kernel_launch(void* const* d_in, const int* in_sizes, int n_in,
                              void* d_out, int out_size, void* d_ws, size_t ws_size,
                              hipStream_t stream) {
    const int*   din   = (const int*)d_in[0];    // decoder_input (N,2)
    const float* enc   = (const float*)d_in[1];  // encoder_out (N,512)
    const float* hlp   = (const float*)d_in[2];  // hyps_log_prob (N,1)
    const float* embed = (const float*)d_in[3];  // embed_table (500,512)
    const float* convw = (const float*)d_in[4];  // conv_w (512,4,2)
    const float* pw    = (const float*)d_in[5];  // proj_w (512,512)
    const float* pb    = (const float*)d_in[6];  // proj_b (512)
    const float* jw    = (const float*)d_in[7];  // joiner_w (500,512)
    const float* jb    = (const float*)d_in[8];  // joiner_b (500)
    // d_in[9] = beam (==4, hardcoded)

    char* ws = (char*)d_ws;
    size_t off = 0;
    unsigned short* pwb  = (unsigned short*)(ws + off); off += (size_t)DIM * DIM * 2;
    unsigned short* jwb  = (unsigned short*)(ws + off); off += (size_t)VPAD * DIM * 2;
    unsigned short* decb = (unsigned short*)(ws + off); off += (size_t)NUM_HYPS * DIM * 2;
    float* cand_val = (float*)(ws + off);               off += (size_t)(NUM_HYPS / MROWS) * 4 * 4;
    int*   cand_idx = (int*)(ws + off);
    float* out = (float*)d_out;

    convert_w_kernel<<<(DIM * DIM) / 256, 256, 0, stream>>>(pw, jw, pwb, jwb);
    frontend_kernel<<<(NUM_HYPS * DIM) / 256, 256, 0, stream>>>(din, embed, convw, decb);
    fused_gemm_topk_kernel<<<NUM_HYPS / MROWS, 256, 0, stream>>>(
        decb, pwb, jwb, pb, enc, jb, hlp, cand_val, cand_idx);
    final_topk_kernel<<<1, 256, 0, stream>>>(cand_val, cand_idx, hlp, out);
}
